// AttnBlock_2207613190437
// MI455X (gfx1250) — matrix operand order repaired and checked by reading, not compiled, not run on hardware
//
#include <hip/hip_runtime.h>
#include <hip/hip_bf16.h>
#include <math.h>

// AttnBlock for MI455X (gfx1250, wave32, WMMA), flash-attention structure.
// Layouts chosen so every WMMA fragment load is contiguous (b128-friendly):
//   h, q, k, o : token-major  [n][c]   (A-frags / K-as-B-frags contiguous)
//   v          : channel-major [c][n]  (V-as-B-frags contiguous)
// Workspace (halves): h/o[4*4096*256] | q | k | v  (32 MB; o aliases h).

#define USE_ASYNC_LDS 1

#define BB 4
#define CC 256
#define NN 4096
#define NGROUPS 32
#define CPG 8

typedef __attribute__((ext_vector_type(16))) _Float16 v16h;
typedef __attribute__((ext_vector_type(8)))  float    v8f;
typedef __attribute__((ext_vector_type(4)))  float    f4_t;

// B-fragment (k = g*16+e): 32 contiguous bytes, 32B-aligned.
__device__ __forceinline__ v16h ldB(const _Float16* p) {
  return *(const v16h*)p;
}
// A-fragment (k = g*16+e): two 16B chunks.
__device__ __forceinline__ v16h ldA(const _Float16* lo, const _Float16* hi) {
  union { f4_t f[2]; v16h h; } u;
  u.f[0] = *(const f4_t*)lo;
  u.f[1] = *(const f4_t*)hi;
  return u.h;
}
__device__ __forceinline__ v8f wmma_f16(const v16h& a, const v16h& b, const v8f& c) {
  return __builtin_amdgcn_wmma_f32_16x16x32_f16(false, a, false, b, (short)0, c,
                                                false, false);
}

#if USE_ASYNC_LDS
__device__ __forceinline__ void async_cp16(unsigned lds_byte, const void* g) {
  asm volatile("global_load_async_to_lds_b128 %0, %1, off"
               :: "v"(lds_byte), "v"(g) : "memory");
}
__device__ __forceinline__ void async_wait0() {
  asm volatile("s_wait_asynccnt 0x0" ::: "memory");
}
#endif

// ---------------------------------------------------------------- GroupNorm
// Output token-major: ht[b][n][c], 16B store per token (8 channels/group).
__global__ __launch_bounds__(256)
void gn_kernel(const float* __restrict__ x, const float* __restrict__ gamma,
               const float* __restrict__ beta, _Float16* __restrict__ ht) {
  const int bg = blockIdx.x;
  const int b = bg / NGROUPS, g = bg % NGROUPS;
  const float* xb = x + ((size_t)b * CC + (size_t)g * CPG) * NN;
  const int tid = threadIdx.x;
  const int TOT = CPG * NN;  // 32768
  float s = 0.f, ss = 0.f;
  for (int i = tid; i < TOT; i += 256) { float v = xb[i]; s += v; ss += v * v; }
  __shared__ float r0[256], r1[256];
  r0[tid] = s; r1[tid] = ss;
  __syncthreads();
  for (int st = 128; st > 0; st >>= 1) {
    if (tid < st) { r0[tid] += r0[tid + st]; r1[tid] += r1[tid + st]; }
    __syncthreads();
  }
  const float mu  = r0[0] / (float)TOT;
  const float var = r1[0] / (float)TOT - mu * mu;
  const float inv = rsqrtf(var + 1e-6f);
  float gm[CPG], bt[CPG];
  #pragma unroll
  for (int c = 0; c < CPG; ++c) {
    gm[c] = gamma[g * CPG + c] * inv;
    bt[c] = beta[g * CPG + c] - mu * gm[c];
  }
  _Float16* hb = ht + (size_t)b * NN * CC + g * CPG;
  for (int t = tid; t < NN; t += 256) {
    alignas(16) _Float16 vals[CPG];
    #pragma unroll
    for (int c = 0; c < CPG; ++c)
      vals[c] = (_Float16)(xb[(size_t)c * NN + t] * gm[c] + bt[c]);
    *(f4_t*)(hb + (size_t)t * CC) = *(const f4_t*)vals;
  }
}

// ------------------------------------------------- QKV projection (3 GEMMs)
// Per WG: 256x128 tile, 8 waves each 64x64 (4x4 WMMA frags).
// q,k written token-major (16B vector epilogue); v channel-major.
__global__ __launch_bounds__(256)
void qkv_kernel(const _Float16* __restrict__ ht,
                const float* __restrict__ wq, const float* __restrict__ bq,
                const float* __restrict__ wk, const float* __restrict__ bk,
                const float* __restrict__ wv, const float* __restrict__ bv,
                _Float16* __restrict__ qo, _Float16* __restrict__ ko,
                _Float16* __restrict__ vo, float qscale) {
  const int nb = blockIdx.x * 128;
  const int b = blockIdx.y;
  const int which = blockIdx.z;
  const float* wmat = (which == 0) ? wq : (which == 1) ? wk : wv;
  const float* bias = (which == 0) ? bq : (which == 1) ? bk : bv;
  _Float16* ob = ((which == 0) ? qo : (which == 1) ? ko : vo) + (size_t)b * CC * NN;
  const float oscale = (which == 0) ? qscale : 1.0f;
  const _Float16* hb = ht + (size_t)b * NN * CC;  // token-major

  __shared__ _Float16 lds_a[256 * 32];  // weights [m][k]  16 KB
  __shared__ _Float16 lds_b[128 * 32];  // h tile  [n][k]   8 KB

  const int tid = threadIdx.x;
  const int wid = tid >> 5, lane = tid & 31;
  const int g = lane >> 4, l16 = lane & 15;
  const int wm = (wid & 3) * 64;
  const int wn = (wid >> 2) * 64;

  v8f acc[4][4] = {};

  for (int kb = 0; kb < CC; kb += 32) {
    {  // A tile: f32 weights -> f16, vector stores
      const f4_t* src = (const f4_t*)(wmat + (size_t)tid * CC + kb);
      alignas(16) _Float16 tmp[32];
      #pragma unroll
      for (int j = 0; j < 8; ++j) {
        f4_t f = src[j];
        tmp[4*j+0] = (_Float16)f[0]; tmp[4*j+1] = (_Float16)f[1];
        tmp[4*j+2] = (_Float16)f[2]; tmp[4*j+3] = (_Float16)f[3];
      }
      f4_t* dst = (f4_t*)(lds_a + tid * 32);
      #pragma unroll
      for (int j = 0; j < 4; ++j) dst[j] = ((const f4_t*)tmp)[j];
    }
    {  // B tile: straight vector copy of token-major rows
      const int n = tid >> 1, part = tid & 1;
      const f4_t* src = (const f4_t*)(hb + (size_t)(nb + n) * CC + kb + part * 16);
      f4_t* dst = (f4_t*)(lds_b + n * 32 + part * 16);
      dst[0] = src[0]; dst[1] = src[1];
    }
    __syncthreads();
    v16h af[4], bf[4];
    #pragma unroll
    for (int i = 0; i < 4; ++i) {
      const _Float16* p = lds_a + (wm + i * 16 + l16) * 32;
      af[i] = ldA(p + g * 16, p + g * 16 + 8);
    }
    #pragma unroll
    for (int j = 0; j < 4; ++j)
      bf[j] = ldB(lds_b + (wn + j * 16 + l16) * 32 + g * 16);
    #pragma unroll
    for (int i = 0; i < 4; ++i)
      #pragma unroll
      for (int j = 0; j < 4; ++j)
        acc[i][j] = wmma_f16(af[i], bf[j], acc[i][j]);
    __syncthreads();
  }

  if (which < 2) {  // token-major, 8 consecutive channels -> 16B store
    #pragma unroll
    for (int i = 0; i < 4; ++i) {
      const int crow0 = wm + i * 16 + g * 8;
      #pragma unroll
      for (int j = 0; j < 4; ++j) {
        const int ncol = nb + wn + j * 16 + l16;
        alignas(16) _Float16 tmp[8];
        #pragma unroll
        for (int r = 0; r < 8; ++r)
          tmp[r] = (_Float16)((acc[i][j][r] + bias[crow0 + r]) * oscale);
        *(f4_t*)(ob + (size_t)ncol * CC + crow0) = *(const f4_t*)tmp;
      }
    }
  } else {  // v: channel-major
    #pragma unroll
    for (int i = 0; i < 4; ++i)
      #pragma unroll
      for (int j = 0; j < 4; ++j) {
        const int ncol = nb + wn + j * 16 + l16;
        #pragma unroll
        for (int r = 0; r < 8; ++r) {
          const int crow = wm + i * 16 + g * 8 + r;
          ob[(size_t)crow * NN + ncol] = (_Float16)(acc[i][j][r] + bias[crow]);
        }
      }
  }
}

// --------------------------------------------------------- Flash attention
// WG = 128 query tokens, 8 waves x 16 rows; keys in tiles of 128.
// LDS: K tile [key][c] 64KB | V tile [c][key] 64KB | P 8x16x128 32KB.
__global__ __launch_bounds__(256)
void attn_kernel(const _Float16* __restrict__ qt, const _Float16* __restrict__ kt,
                 const _Float16* __restrict__ vc, _Float16* __restrict__ ot) {
  extern __shared__ _Float16 smem[];
  _Float16* ldsK = smem;                    // [key=128][c=256]
  _Float16* ldsV = smem + 128 * 256;        // [c=256][key=128]
  _Float16* ldsP = smem + 2 * 128 * 256;    // [wave=8][m=16][key=128]

  const int qb = blockIdx.x * 128;
  const int b = blockIdx.y;
  const _Float16* qB = qt + (size_t)b * NN * CC;
  const _Float16* kB = kt + (size_t)b * NN * CC;
  const _Float16* vB = vc + (size_t)b * CC * NN;
  _Float16* oB = ot + (size_t)b * NN * CC;

  const int tid = threadIdx.x;
  const int wid = tid >> 5, lane = tid & 31;
  const int g = lane >> 4, l16 = lane & 15;
  const int qrow0 = wid * 16;
  _Float16* myP = ldsP + wid * (16 * 128);

  // Q A-fragments straight from global token-major rows (b128 chunks)
  v16h aq[8];
  {
    const _Float16* qrow = qB + (size_t)(qb + qrow0 + l16) * CC;
    #pragma unroll
    for (int cs = 0; cs < 8; ++cs)
      aq[cs] = ldA(qrow + cs * 32 + g * 16, qrow + cs * 32 + g * 16 + 8);
  }

  v8f oacc[16] = {};   // 16 query rows x 256 channels
  float rm[8], rl[8];  // per-row running max / denom (rows m = g*8+r)
  #pragma unroll
  for (int r = 0; r < 8; ++r) { rm[r] = -__builtin_inff(); rl[r] = 0.f; }

  for (int kb = 0; kb < NN; kb += 128) {
    {  // stage K tile (2 threads/row, 256B each) and V tile (1 thread/row)
      const int krow = tid >> 1, kpart = tid & 1;
      const _Float16* ksrc = kB + (size_t)(kb + krow) * CC + kpart * 128;
      const _Float16* vsrc = vB + (size_t)tid * NN + kb;
#if USE_ASYNC_LDS
      const unsigned kdst = (unsigned)(size_t)(ldsK + krow * 256 + kpart * 128);
      const unsigned vdst = (unsigned)(size_t)(ldsV + tid * 128);
      #pragma unroll
      for (int i = 0; i < 16; ++i) {
        async_cp16(kdst + i * 16, (const char*)ksrc + i * 16);
        async_cp16(vdst + i * 16, (const char*)vsrc + i * 16);
      }
      async_wait0();
#else
      f4_t* dk = (f4_t*)(ldsK + krow * 256 + kpart * 128);
      f4_t* dv = (f4_t*)(ldsV + tid * 128);
      const f4_t* sk = (const f4_t*)ksrc;
      const f4_t* sv = (const f4_t*)vsrc;
      #pragma unroll
      for (int i = 0; i < 16; ++i) { dk[i] = sk[i]; dv[i] = sv[i]; }
#endif
    }
    if (kb + 128 < NN) {  // global_prefetch_b8 of next tiles
      __builtin_prefetch(kB + (size_t)(kb + 128 + (tid >> 1)) * CC, 0, 1);
      __builtin_prefetch(vB + (size_t)tid * NN + kb + 128, 0, 1);
    }
    __syncthreads();

    // S tile: 16 rows x 128 keys (scale*log2e folded into Q)
    v8f s[8] = {};
    #pragma unroll
    for (int nt = 0; nt < 8; ++nt) {
      const _Float16* krow = ldsK + (nt * 16 + l16) * 256 + g * 16;
      #pragma unroll
      for (int cs = 0; cs < 8; ++cs)
        s[nt] = wmma_f16(aq[cs], ldB(krow + cs * 32), s[nt]);
    }

    // online softmax: row max over 8 frags + 16-lane shfl reduction
    float tmax[8];
    #pragma unroll
    for (int r = 0; r < 8; ++r) {
      float m = s[0][r];
      #pragma unroll
      for (int nt = 1; nt < 8; ++nt) m = fmaxf(m, s[nt][r]);
      #pragma unroll
      for (int off = 1; off < 16; off <<= 1) m = fmaxf(m, __shfl_xor(m, off, 32));
      tmax[r] = m;
    }
    float alpha[8];
    #pragma unroll
    for (int r = 0; r < 8; ++r) {
      const float nm = fmaxf(rm[r], tmax[r]);
      alpha[r] = exp2f(rm[r] - nm);
      rm[r] = nm;
      rl[r] *= alpha[r];
    }
    #pragma unroll
    for (int nt = 0; nt < 16; ++nt)
      #pragma unroll
      for (int r = 0; r < 8; ++r) oacc[nt][r] *= alpha[r];

    float rs[8] = {0, 0, 0, 0, 0, 0, 0, 0};
    #pragma unroll
    for (int nt = 0; nt < 8; ++nt)
      #pragma unroll
      for (int r = 0; r < 8; ++r) {
        const float p = exp2f(s[nt][r] - rm[r]);
        rs[r] += p;
        myP[(g * 8 + r) * 128 + nt * 16 + l16] = (_Float16)p;  // C->A relayout
      }
    #pragma unroll
    for (int r = 0; r < 8; ++r) {
      float t = rs[r];
      #pragma unroll
      for (int off = 1; off < 16; off <<= 1) t += __shfl_xor(t, off, 32);
      rl[r] += t;
    }

    // O += P * V^T
    v16h ap[4];
    #pragma unroll
    for (int ks = 0; ks < 4; ++ks) {
      const _Float16* p = myP + l16 * 128 + ks * 32;
      ap[ks] = ldA(p + g * 16, p + g * 16 + 8);
    }
    #pragma unroll
    for (int nt = 0; nt < 16; ++nt) {
      const _Float16* vrow = ldsV + (nt * 16 + l16) * 128 + g * 16;
      #pragma unroll
      for (int ks = 0; ks < 4; ++ks)
        oacc[nt] = wmma_f16(ap[ks], ldB(vrow + ks * 32), oacc[nt]);
    }
    __syncthreads();
  }

  float invl[8];
  #pragma unroll
  for (int r = 0; r < 8; ++r) invl[r] = 1.0f / rl[r];
  #pragma unroll
  for (int r = 0; r < 8; ++r) {
    const int tok = qb + qrow0 + g * 8 + r;
    _Float16* orow = oB + (size_t)tok * CC;
    #pragma unroll
    for (int nt = 0; nt < 16; ++nt)
      orow[nt * 16 + l16] = (_Float16)(oacc[nt][r] * invl[r]);
  }
}

// ------------------------------------- output projection + residual (f32)
__global__ __launch_bounds__(256)
void proj_kernel(const _Float16* __restrict__ ot, const float* __restrict__ wo,
                 const float* __restrict__ bo, const float* __restrict__ x,
                 float* __restrict__ out) {
  const int nb = blockIdx.x * 128;
  const int b = blockIdx.y;
  const _Float16* obuf = ot + (size_t)b * NN * CC;  // token-major

  __shared__ _Float16 lds_a[256 * 32];
  __shared__ _Float16 lds_b[128 * 32];

  const int tid = threadIdx.x;
  const int wid = tid >> 5, lane = tid & 31;
  const int g = lane >> 4, l16 = lane & 15;
  const int wm = (wid & 3) * 64;
  const int wn = (wid >> 2) * 64;

  v8f acc[4][4] = {};

  for (int kb = 0; kb < CC; kb += 32) {
    {
      const f4_t* src = (const f4_t*)(wo + (size_t)tid * CC + kb);
      alignas(16) _Float16 tmp[32];
      #pragma unroll
      for (int j = 0; j < 8; ++j) {
        f4_t f = src[j];
        tmp[4*j+0] = (_Float16)f[0]; tmp[4*j+1] = (_Float16)f[1];
        tmp[4*j+2] = (_Float16)f[2]; tmp[4*j+3] = (_Float16)f[3];
      }
      f4_t* dst = (f4_t*)(lds_a + tid * 32);
      #pragma unroll
      for (int j = 0; j < 4; ++j) dst[j] = ((const f4_t*)tmp)[j];
    }
    {
      const int n = tid >> 1, part = tid & 1;
      const f4_t* src = (const f4_t*)(obuf + (size_t)(nb + n) * CC + kb + part * 16);
      f4_t* dst = (f4_t*)(lds_b + n * 32 + part * 16);
      dst[0] = src[0]; dst[1] = src[1];
    }
    __syncthreads();
    v16h af[4], bf[4];
    #pragma unroll
    for (int i = 0; i < 4; ++i) {
      const _Float16* p = lds_a + (wm + i * 16 + l16) * 32;
      af[i] = ldA(p + g * 16, p + g * 16 + 8);
    }
    #pragma unroll
    for (int j = 0; j < 4; ++j)
      bf[j] = ldB(lds_b + (wn + j * 16 + l16) * 32 + g * 16);
    #pragma unroll
    for (int i = 0; i < 4; ++i)
      #pragma unroll
      for (int j = 0; j < 4; ++j)
        acc[i][j] = wmma_f16(af[i], bf[j], acc[i][j]);
    __syncthreads();
  }
  #pragma unroll
  for (int i = 0; i < 4; ++i)
    #pragma unroll
    for (int j = 0; j < 4; ++j) {
      const int ncol = nb + wn + j * 16 + l16;
      #pragma unroll
      for (int r = 0; r < 8; ++r) {
        const int crow = wm + i * 16 + g * 8 + r;
        const size_t idx = ((size_t)b * CC + crow) * NN + ncol;
        out[idx] = x[idx] + acc[i][j][r] + bo[crow];
      }
    }
}

extern "C" void kernel_launch(void* const* d_in, const int* in_sizes, int n_in,
                              void* d_out, int out_size, void* d_ws, size_t ws_size,
                              hipStream_t stream) {
  const float* x     = (const float*)d_in[0];
  const float* gamma = (const float*)d_in[1];
  const float* beta  = (const float*)d_in[2];
  const float* wq    = (const float*)d_in[3];
  const float* bq    = (const float*)d_in[4];
  const float* wk    = (const float*)d_in[5];
  const float* bk    = (const float*)d_in[6];
  const float* wv    = (const float*)d_in[7];
  const float* bv    = (const float*)d_in[8];
  const float* wo    = (const float*)d_in[9];
  const float* bo    = (const float*)d_in[10];
  float* out = (float*)d_out;

  const size_t n1 = (size_t)BB * CC * NN;
  _Float16* h  = (_Float16*)d_ws;  // token-major
  _Float16* qb = h + n1;           // token-major
  _Float16* kb = qb + n1;          // token-major
  _Float16* vb = kb + n1;          // channel-major
  _Float16* ob = h;                // attention output aliases h (token-major)

  // fold 1/sqrt(C) and log2(e) into Q so softmax uses exp2 (v_exp_f32)
  const float qscale = (1.0f / 16.0f) * 1.44269504088896340736f;

  gn_kernel<<<dim3(BB * NGROUPS), 256, 0, stream>>>(x, gamma, beta, h);
  qkv_kernel<<<dim3(NN / 128, BB, 3), 256, 0, stream>>>(
      h, wq, bq, wk, bk, wv, bv, qb, kb, vb, qscale);
  attn_kernel<<<dim3(NN / 128, BB), 256, 160 * 1024, stream>>>(qb, kb, vb, ob);
  proj_kernel<<<dim3(NN / 128, BB), 256, 0, stream>>>(ob, wo, bo, x, out);
  (void)in_sizes; (void)n_in; (void)out_size; (void)ws_size;
}
